// FixFeatureAttention_1529008358153
// MI455X (gfx1250) — matrix operand order repaired and checked by reading, not compiled, not run on hardware
//
#include <hip/hip_runtime.h>
#include <hip/hip_bf16.h>

typedef __attribute__((ext_vector_type(16))) __bf16 v16bf;
typedef __attribute__((ext_vector_type(8)))  __bf16 v8bf;
typedef __attribute__((ext_vector_type(8)))  float  v8f;
typedef __attribute__((ext_vector_type(4)))  int    i32x4;

typedef __attribute__((address_space(1))) i32x4 g_i32x4;  // global (prints __device__)
typedef __attribute__((address_space(3))) i32x4 l_i32x4;  // LDS (prints __shared__)

#define HEADS 4
#define NB    16
#define NC    256
#define HW    1024
#define INNER 128
#define DHEAD 32

// gfx1250 async global->LDS path (ASYNCcnt-tracked), guarded so a missing
// builtin falls back to the plain load/ds_store copy.
#if defined(__has_builtin)
#if __has_builtin(__builtin_amdgcn_global_load_async_to_lds_b128)
#define HAVE_ASYNC_LDS 1
#endif
#endif

static __device__ __forceinline__ void wait_asynccnt0() {
#if defined(__has_builtin)
#if __has_builtin(__builtin_amdgcn_s_wait_asynccnt)
  __builtin_amdgcn_s_wait_asynccnt(0);
  return;
#else
  asm volatile("s_wait_asynccnt 0x0" ::: "memory");
#endif
#endif
}

static __device__ __forceinline__ v16bf cat8(v8bf a, v8bf b) {
  v16bf r;
#pragma unroll
  for (int i = 0; i < 8; ++i) { r[i] = a[i]; r[i + 8] = b[i]; }
  return r;
}
static __device__ __forceinline__ v8f vzero8() {
  v8f r = {0.f, 0.f, 0.f, 0.f, 0.f, 0.f, 0.f, 0.f};
  return r;
}
static __device__ __forceinline__ v8f wmma_bf16(v16bf a, v16bf b, v8f c) {
  // (neg_a, A, neg_b, B, c_mod, C, reuse_a, reuse_b)
  return __builtin_amdgcn_wmma_f32_16x16x32_bf16(false, a, false, b, (short)0, c,
                                                 false, false);
}

// ---------------------------------------------------------------------------
// Kernel 1: qkv = w_qkv[384,256] @ x[b,256,1024], stored bf16 in WMMA-friendly
// layouts: qT [b][h][i][d] (scaled by d^-0.5), kT [b][h][j][d], v [b][h][d][j].
// One wave per 16x16 output tile; K-loop of 8 x (16x16x32 bf16 WMMA).
// ---------------------------------------------------------------------------
__global__ void k_qkv(const float* __restrict__ x, const float* __restrict__ wqkv,
                      __bf16* __restrict__ qT, __bf16* __restrict__ kT,
                      __bf16* __restrict__ vB) {
  const int lane = threadIdx.x & 31;
  const int ln = lane & 15, lh = lane >> 4;
  const int ntile = blockIdx.x;   // 64
  const int otile = blockIdx.y;   // 24
  const int b = blockIdx.z;       // 16
  const int n = ntile * 16 + ln;
  const int orow = otile * 16 + ln;

  v8f acc = vzero8();
#pragma unroll
  for (int kk = 0; kk < 8; ++kk) {
    const int cb = kk * 32;
    v16bf a, bb;
    const float* wr = wqkv + (size_t)orow * NC + cb;
#pragma unroll
    for (int e = 0; e < 8; ++e) {
      a[e]     = (__bf16)wr[lh * 16 + e];       // K = lh*16 + e
      a[e + 8] = (__bf16)wr[lh * 16 + 8 + e];   // K = lh*16 + 8 + e
    }
    const float* xc = x + (size_t)b * NC * HW + (size_t)(cb + lh * 16) * HW + n;
    if (kk < 7) __builtin_prefetch(xc + (size_t)32 * HW, 0, 1);  // next c-slab
#pragma unroll
    for (int e = 0; e < 16; ++e) bb[e] = (__bf16)xc[(size_t)e * HW];
    acc = wmma_bf16(a, bb, acc);
  }

  const float scale = 0.17677669529663687f;  // 1/sqrt(32)
#pragma unroll
  for (int r = 0; r < 8; ++r) {
    const int og = otile * 16 + r + 8 * lh;  // output channel row of this value
    const float val = acc[r];
    if (otile < 8) {           // q rows 0..127
      const int hh = og >> 5, d = og & 31;
      qT[(((size_t)b * HEADS + hh) * HW + n) * DHEAD + d] = (__bf16)(val * scale);
    } else if (otile < 16) {   // k rows 128..255
      const int oc = og - 128;
      const int hh = oc >> 5, d = oc & 31;
      kT[(((size_t)b * HEADS + hh) * HW + n) * DHEAD + d] = (__bf16)val;
    } else {                   // v rows 256..383
      const int oc = og - 256;
      const int hh = oc >> 5, d = oc & 31;
      vB[(((size_t)b * HEADS + hh) * DHEAD + d) * HW + n] = (__bf16)val;
    }
  }
}

// ---------------------------------------------------------------------------
// Kernel 2: bilinear upsample pos_bias [1,4,16,16] -> pos [4,1024]
// (half-pixel centers, edge clamp == torch align_corners=False / jax resize)
// ---------------------------------------------------------------------------
__global__ void k_pos(const float* __restrict__ pb, float* __restrict__ pos) {
  const int idx = blockIdx.x * blockDim.x + threadIdx.x;
  if (idx >= HEADS * HW) return;
  const int h = idx >> 10, p = idx & 1023;
  const int yy = p >> 5, xx = p & 31;
  const float sy = (yy + 0.5f) * 0.5f - 0.5f;
  const float sx = (xx + 0.5f) * 0.5f - 0.5f;
  const int y0 = (int)floorf(sy), x0 = (int)floorf(sx);
  const float wy = sy - (float)y0, wx = sx - (float)x0;
  const int y0c = max(y0, 0), y1c = min(y0 + 1, 15);
  const int x0c = max(x0, 0), x1c = min(x0 + 1, 15);
  const float* s = pb + h * 256;
  const float v00 = s[y0c * 16 + x0c], v01 = s[y0c * 16 + x1c];
  const float v10 = s[y1c * 16 + x0c], v11 = s[y1c * 16 + x1c];
  const float v0 = v00 + (v01 - v00) * wx;
  const float v1 = v10 + (v11 - v10) * wx;
  pos[idx] = v0 + (v1 - v0) * wy;
}

// ---------------------------------------------------------------------------
// Kernel 3: fused attention per (b,h). K [1024][32] and V [32][1024] (bf16)
// live entirely in LDS (128KB of the 320KB WGP LDS) + pos row (4KB).
// K/V are streamed in with GLOBAL_LOAD_ASYNC_TO_LDS_B128 (ASYNCcnt) when the
// toolchain exposes it. Each wave owns 16 queries at a time; S is computed
// TRANSPOSED (keys=M, queries=N) so online-softmax stats reduce per-lane +
// one shfl_xor(16). P needs no lane exchange: its C-layout key order is used as the B-layout.
// ---------------------------------------------------------------------------
__global__ void k_attn(const __bf16* __restrict__ qT, const __bf16* __restrict__ kT,
                       const __bf16* __restrict__ vB, const float* __restrict__ pos,
                       __bf16* __restrict__ yT) {
  extern __shared__ char smem[];
  __bf16* Kls = (__bf16*)smem;                  // [1024][32]  64KB
  __bf16* Vls = (__bf16*)(smem + 65536);        // [32][1024]  64KB
  float* posS = (float*)(smem + 131072);        // [1024]       4KB

  const int tid = threadIdx.x;
  const int qchunk = blockIdx.x;  // 8  (8 query tiles each)
  const int h = blockIdx.y;       // 4
  const int b = blockIdx.z;       // 16
  const size_t bh = (size_t)b * HEADS + h;

  const __bf16* kg = kT + bh * HW * DHEAD;
  const __bf16* vg = vB + bh * DHEAD * HW;
#if defined(HAVE_ASYNC_LDS)
  {
    // 4096 x 16B granules each for K and V, direct-to-LDS (no VGPR staging).
    for (int i = tid; i < HW * DHEAD / 8; i += blockDim.x) {
      __builtin_amdgcn_global_load_async_to_lds_b128(
          (g_i32x4*)(kg + (size_t)i * 8), (l_i32x4*)(Kls + (size_t)i * 8), 0, 0);
      __builtin_amdgcn_global_load_async_to_lds_b128(
          (g_i32x4*)(vg + (size_t)i * 8), (l_i32x4*)(Vls + (size_t)i * 8), 0, 0);
    }
    const float4* ps = (const float4*)(pos + h * HW);
    float4* pd = (float4*)posS;
    for (int i = tid; i < HW / 4; i += blockDim.x) pd[i] = ps[i];
    wait_asynccnt0();
  }
#else
  {
    const ulonglong2* ks = (const ulonglong2*)kg;
    const ulonglong2* vs = (const ulonglong2*)vg;
    ulonglong2* kd = (ulonglong2*)Kls;
    ulonglong2* vd = (ulonglong2*)Vls;
    for (int i = tid; i < HW * DHEAD * 2 / 16; i += blockDim.x) {
      kd[i] = ks[i];
      vd[i] = vs[i];
    }
    const float4* ps = (const float4*)(pos + h * HW);
    float4* pd = (float4*)posS;
    for (int i = tid; i < HW / 4; i += blockDim.x) pd[i] = ps[i];
  }
#endif
  __syncthreads();

  const int wave = tid >> 5, lane = tid & 31;
  const int ln = lane & 15, lh = lane >> 4;

  for (int ql = 0; ql < 2; ++ql) {
    const int qtile = qchunk * 8 + wave * 2 + ql;
    const int qbase = qtile * 16;

    // Q B-fragment: lane column = query qbase+ln, K-dim = d (lh*8 + e, 16 + lh*8 + e)
    const __bf16* qrow = qT + (bh * HW + qbase + ln) * DHEAD + lh * 8;
    const v16bf bq = cat8(*(const v8bf*)qrow, *(const v8bf*)(qrow + 16));
    const float pi = posS[qbase + ln];

    float run_max = -3.0e38f, run_sum = 0.0f;
    v8f o0 = vzero8(), o1 = vzero8();

    for (int kb = 0; kb < HW; kb += 32) {
      // K A-fragments: row m = key, K-dim = d
      const __bf16* kr0 = Kls + (size_t)(kb + ln) * DHEAD + lh * 8;
      const __bf16* kr1 = Kls + (size_t)(kb + 16 + ln) * DHEAD + lh * 8;
      const v16bf ak0 = cat8(*(const v8bf*)kr0, *(const v8bf*)(kr0 + 16));
      const v16bf ak1 = cat8(*(const v8bf*)kr1, *(const v8bf*)(kr1 + 16));
      v8f c0 = wmma_bf16(ak0, bq, vzero8());  // S^T tile: m=key kb..kb+15
      v8f c1 = wmma_bf16(ak1, bq, vzero8());  // m=key kb+16..kb+31

      float tmax = -3.0e38f;
#pragma unroll
      for (int r = 0; r < 8; ++r) {
        c0[r] += pi + posS[kb + lh * 8 + r];
        c1[r] += pi + posS[kb + 16 + lh * 8 + r];
        tmax = fmaxf(tmax, fmaxf(c0[r], c1[r]));
      }
      tmax = fmaxf(tmax, __shfl_xor(tmax, 16, 32));
      const float nm = fmaxf(run_max, tmax);
      const float corr = __expf(run_max - nm);
      float ts = 0.0f;
#pragma unroll
      for (int r = 0; r < 8; ++r) {
        o0[r] *= corr;
        o1[r] *= corr;
        c0[r] = __expf(c0[r] - nm);
        c1[r] = __expf(c1[r] - nm);
        ts += c0[r] + c1[r];
      }
      run_sum = run_sum * corr + ts + __shfl_xor(ts, 16, 32);
      run_max = nm;

      // Repack P (C layout) -> B fragment (32 keys x 16 queries)
      v16bf bp;
#pragma unroll
      for (int r = 0; r < 8; ++r) {
        const float pc0 = __shfl_xor(c0[r], 16, 32);
        const float pc1 = __shfl_xor(c1[r], 16, 32);
        bp[r]     = (__bf16)c0[r];  // keys kb + lh*8 + r
        bp[r + 8] = (__bf16)c1[r];  // keys kb + 16 + lh*8 + r
      }

      // V A-fragments: row m = d, K-dim = key
      const __bf16* vr0 = Vls + (size_t)ln * HW + kb + lh * 8;
      const __bf16* vr1 = Vls + (size_t)(16 + ln) * HW + kb + lh * 8;
      const v16bf av0 = cat8(*(const v8bf*)vr0, *(const v8bf*)(vr0 + 16));
      const v16bf av1 = cat8(*(const v8bf*)vr1, *(const v8bf*)(vr1 + 16));
      o0 = wmma_bf16(av0, bp, o0);  // d = 0..15
      o1 = wmma_bf16(av1, bp, o1);  // d = 16..31
    }

    const float inv = 1.0f / run_sum;
    v8bf w0, w1;
#pragma unroll
    for (int r = 0; r < 8; ++r) {
      w0[r] = (__bf16)(o0[r] * inv);
      w1[r] = (__bf16)(o1[r] * inv);
    }
    // D layout: lane holds query qbase+ln, d = vgpr + 8*lh (+16 for o1)
    __bf16* yrow = yT + ((size_t)b * HW + qbase + ln) * INNER + h * DHEAD;
    *(v8bf*)(yrow + lh * 8) = w0;
    *(v8bf*)(yrow + 16 + lh * 8) = w1;
  }
}

// ---------------------------------------------------------------------------
// Kernel 4: out = LN_channel(x + w_proj @ y + b_proj) * gamma + beta.
// One block per (b, 16-column i-tile): 4 waves x 4 o-tiles x 4 WMMAs build the
// full 256-channel column block in LDS, then a two-stage LDS reduction does
// the channel LayerNorm (biased variance, eps=1e-5).
// ---------------------------------------------------------------------------
__global__ void k_proj_ln(const float* __restrict__ x, const float* __restrict__ wproj,
                          const float* __restrict__ bproj, const __bf16* __restrict__ yT,
                          const float* __restrict__ gamma, const float* __restrict__ beta,
                          float* __restrict__ out) {
  __shared__ float zbuf[NC][16];
  __shared__ float psum[8][16];
  __shared__ float psq[8][16];
  __shared__ float smean[16];
  __shared__ float srinv[16];

  const int tid = threadIdx.x;
  const int itile = blockIdx.x;  // 64
  const int b = blockIdx.y;      // 16
  const int wave = tid >> 5, lane = tid & 31;
  const int ln = lane & 15, lh = lane >> 4;
  const int ibase = itile * 16;

#pragma unroll
  for (int t = 0; t < 4; ++t) {
    const int ot = wave * 4 + t;  // 16 o-tiles cover 256 channels
    const int orow = ot * 16 + ln;
    v8f acc = vzero8();
#pragma unroll
    for (int ks = 0; ks < 4; ++ks) {
      v16bf a;
      const float* wr = wproj + (size_t)orow * INNER + ks * 32;
#pragma unroll
      for (int e = 0; e < 8; ++e) {
        a[e]     = (__bf16)wr[lh * 8 + e];
        a[e + 8] = (__bf16)wr[16 + lh * 8 + e];
      }
      const __bf16* yr =
          yT + ((size_t)b * HW + ibase + ln) * INNER + ks * 32 + lh * 8;
      const v16bf bb = cat8(*(const v8bf*)yr, *(const v8bf*)(yr + 16));
      acc = wmma_bf16(a, bb, acc);
    }
#pragma unroll
    for (int r = 0; r < 8; ++r) {
      const int o = ot * 16 + r + 8 * lh;
      const float z = acc[r] + bproj[o] + x[((size_t)b * NC + o) * HW + ibase + ln];
      zbuf[o][ln] = z;
    }
  }
  __syncthreads();

  const int ii = tid & 15, chunk = tid >> 4;  // 8 chunks x 32 channels
  float s = 0.f, sq = 0.f;
  for (int o = chunk * 32; o < chunk * 32 + 32; ++o) {
    const float z = zbuf[o][ii];
    s += z;
    sq += z * z;
  }
  psum[chunk][ii] = s;
  psq[chunk][ii] = sq;
  __syncthreads();
  if (tid < 16) {
    float S = 0.f, SQ = 0.f;
    for (int c2 = 0; c2 < 8; ++c2) {
      S += psum[c2][tid];
      SQ += psq[c2][tid];
    }
    const float mean = S * (1.0f / 256.0f);
    const float var = SQ * (1.0f / 256.0f) - mean * mean;
    smean[tid] = mean;
    srinv[tid] = rsqrtf(var + 1e-5f);
  }
  __syncthreads();
  const float mean = smean[ii], rinv = srinv[ii];
  for (int o = chunk * 32; o < chunk * 32 + 32; ++o) {
    const float v = (zbuf[o][ii] - mean) * rinv * gamma[o] + beta[o];
    out[((size_t)b * NC + o) * HW + ibase + ii] = v;
  }
}

// ---------------------------------------------------------------------------
extern "C" void kernel_launch(void* const* d_in, const int* in_sizes, int n_in,
                              void* d_out, int out_size, void* d_ws, size_t ws_size,
                              hipStream_t stream) {
  const float* x     = (const float*)d_in[0];  // [16,256,32,32]
  const float* wqkv  = (const float*)d_in[1];  // [384,256]
  const float* wproj = (const float*)d_in[2];  // [256,128]
  const float* bproj = (const float*)d_in[3];  // [256]
  const float* pb    = (const float*)d_in[4];  // [1,4,16,16]
  const float* gamma = (const float*)d_in[5];  // [256]
  const float* beta  = (const float*)d_in[6];  // [256]
  float* out = (float*)d_out;                  // [16,256,32,32]

  char* ws = (char*)d_ws;
  __bf16* qT = (__bf16*)ws;                         // 4 MB  [16][4][1024][32]
  __bf16* kT = (__bf16*)(ws + (size_t)(4 << 20));   // 4 MB  [16][4][1024][32]
  __bf16* vB = (__bf16*)(ws + (size_t)(8 << 20));   // 4 MB  [16][4][32][1024]
  float* pos = (float*)(ws + (size_t)(12 << 20));   // 16 KB [4][1024]
  __bf16* yT = (__bf16*)(ws + (size_t)(12 << 20) + (1 << 16));  // 4 MB [16][1024][128]

  k_qkv<<<dim3(64, 24, 16), 32, 0, stream>>>(x, wqkv, qT, kT, vB);
  k_pos<<<dim3(16), 256, 0, stream>>>(pb, pos);
  const size_t smem = 65536 * 2 + 4096;  // K + V + pos in LDS (132 KB of 320 KB)
  k_attn<<<dim3(8, 4, 16), 128, smem, stream>>>(qT, kT, vB, pos, yT);
  k_proj_ln<<<dim3(64, 16), 128, 0, stream>>>(x, wproj, bproj, yT, gamma, beta, out);
}
